// TransformerBlock_21019569947121
// MI455X (gfx1250) — matrix operand order repaired and checked by reading, not compiled, not run on hardware
//
#include <hip/hip_runtime.h>

// Problem constants (match reference)
#define B_   4
#define S_   2048
#define DM_  1024
#define H_   16
#define HD_  64
#define FF_  4096
#define EPS_ 1e-5f

// CDNA5 WMMA vector types (wave32)
typedef __attribute__((ext_vector_type(16))) __bf16 v16bf;
typedef __attribute__((ext_vector_type(8)))  __bf16 bf16x8;
typedef __attribute__((ext_vector_type(4)))  __bf16 bf16x4;
typedef __attribute__((ext_vector_type(8)))  float  v8f;
typedef __attribute__((ext_vector_type(4)))  int    i32x4;

union Frag16 { v16bf v; bf16x8 h8[2]; };

#ifndef __has_builtin
#define __has_builtin(x) 0
#endif
#if __has_builtin(__builtin_amdgcn_global_load_async_to_lds_b128)
#define HAVE_ASYNC_LDS 1
#else
#define HAVE_ASYNC_LDS 0
#endif
#if __has_builtin(__builtin_amdgcn_s_wait_asynccnt)
#define HAVE_WAIT_ASYNC 1
#else
#define HAVE_WAIT_ASYNC 0
#endif

// global (b128) -> LDS copy; CDNA5 async-to-LDS path when available.
// Builtin signature (from hipcc diagnostic): (v4i AS1*, v4i AS3*, imm, imm).
__device__ __forceinline__ void copy_b128_g2l(const __bf16* gp, __bf16* lp) {
#if HAVE_ASYNC_LDS
  __builtin_amdgcn_global_load_async_to_lds_b128(
      (__attribute__((address_space(1))) i32x4*)gp,
      (__attribute__((address_space(3))) i32x4*)lp, 0, 0);
#else
  *(bf16x8*)lp = *(const bf16x8*)gp;
#endif
}
__device__ __forceinline__ void wait_async_lds() {
#if HAVE_ASYNC_LDS
#if HAVE_WAIT_ASYNC
  __builtin_amdgcn_s_wait_asynccnt(0);
#else
  asm volatile("s_wait_asynccnt 0x0" ::: "memory");
#endif
#endif
}

__device__ __forceinline__ float gelu_tanh(float x) {
  float x3 = x * x * x;
  return 0.5f * x * (1.0f + tanhf(0.7978845608f * (x + 0.044715f * x3)));
}

// ---------------------------------------------------------------------------
// Elementwise f32 -> bf16 (4 per thread, b128 in / b64 out)
// ---------------------------------------------------------------------------
__global__ __launch_bounds__(256) void to_bf16_kernel(
    const float* __restrict__ in, __bf16* __restrict__ out) {
  const size_t i = ((size_t)blockIdx.x * 256 + threadIdx.x) * 4;
  const float4 f = *(const float4*)(in + i);
  bf16x4 o = {(__bf16)f.x, (__bf16)f.y, (__bf16)f.z, (__bf16)f.w};
  *(bf16x4*)(out + i) = o;
}

// ---------------------------------------------------------------------------
// Transpose + convert: in (R,C) f32 row-major -> out (C,R) bf16 row-major
// ---------------------------------------------------------------------------
__global__ __launch_bounds__(256) void transpose_to_bf16(
    const float* __restrict__ in, __bf16* __restrict__ out, int R, int C) {
  __shared__ float tile[32][33];
  const int nt = blockIdx.x * 32;  // col tile in input
  const int kt = blockIdx.y * 32;  // row tile in input
  const int tx = threadIdx.x & 31, ty = threadIdx.x >> 5;  // 32 x 8
  #pragma unroll
  for (int r = ty; r < 32; r += 8)
    tile[r][tx] = in[(size_t)(kt + r) * C + (nt + tx)];
  __syncthreads();
  #pragma unroll
  for (int r = ty; r < 32; r += 8)
    out[(size_t)(nt + r) * R + (kt + tx)] = (__bf16)tile[tx][r];
}

// ---------------------------------------------------------------------------
// GEMM: Y = act(A @ Wt^T + bias)
// A:  (M,K) bf16 row-major. Wt: (N,K) bf16 row-major (pre-transposed weight).
// Block tile 128x128, BK=32, 8 waves each computing 32x64 (8 WMMAs/k-step).
// Double-buffered LDS, staged with GLOBAL_LOAD_ASYNC_TO_LDS_B128 when present.
// Fragment layouts used here (contiguous K, same for A and B, wave32):
//   A frag: row = lane%16, K = (lane/16)*16 + {0..15}
//   B frag: col = lane%16, K = (lane/16)*16 + {0..15}
//   C/D:    rows = (lane/16)*8 + r, col = lane%16
// ---------------------------------------------------------------------------
template <int ACT, int OUT_BF16>
__global__ __launch_bounds__(256) void gemm_bf16(
    const __bf16* __restrict__ A, const __bf16* __restrict__ Wt,
    const float* __restrict__ bias, float* __restrict__ Yf,
    __bf16* __restrict__ Yb, int M, int N, int K) {
  constexpr int BM = 128, BN = 128, BK = 32;
  __shared__ __align__(16) __bf16 sA[2][BM * BK];
  __shared__ __align__(16) __bf16 sB[2][BN * BK];

  const int tid  = threadIdx.x;
  const int lane = tid & 31;
  const int wave = tid >> 5;
  const int wm   = wave >> 1;  // 0..3 : 32-row slice
  const int wn   = wave & 1;   // 0..1 : 64-col slice
  const int m0   = blockIdx.x * BM;
  const int n0   = blockIdx.y * BN;
  const int lrow = lane & 15;
  const int lhi  = lane >> 4;
  const int kbA  = lhi * 16;

  v8f acc[2][4] = {};

  auto stage = [&](int buf, int k0) {
    #pragma unroll
    for (int i = 0; i < 2; ++i) {
      const int idx = tid + i * 256;              // 0..511 b128 chunks
      const int row = idx >> 2, c = (idx & 3) * 8;
      copy_b128_g2l(A  + (size_t)(m0 + row) * K + k0 + c, &sA[buf][row * BK + c]);
      copy_b128_g2l(Wt + (size_t)(n0 + row) * K + k0 + c, &sB[buf][row * BK + c]);
    }
  };

  stage(0, 0);
  int cur = 0;
  for (int k0 = 0; k0 < K; k0 += BK) {
    wait_async_lds();
    __syncthreads();
    if (k0 + BK < K) stage(cur ^ 1, k0 + BK);

    Frag16 af[2], wf[4];
    #pragma unroll
    for (int t = 0; t < 2; ++t) {
      const int arow = wm * 32 + t * 16 + lrow;
      af[t].h8[0] = *(const bf16x8*)&sA[cur][arow * BK + kbA];
      af[t].h8[1] = *(const bf16x8*)&sA[cur][arow * BK + kbA + 8];
    }
    #pragma unroll
    for (int t = 0; t < 4; ++t) {
      const int bcol = wn * 64 + t * 16 + lrow;
      wf[t].v = *(const v16bf*)&sB[cur][bcol * BK + lhi * 16];
    }
    #pragma unroll
    for (int i = 0; i < 2; ++i)
      #pragma unroll
      for (int j = 0; j < 4; ++j)
        acc[i][j] = __builtin_amdgcn_wmma_f32_16x16x32_bf16(
            false, af[i].v, false, wf[j].v, (short)0, acc[i][j], false, false);
    cur ^= 1;
  }

  #pragma unroll
  for (int j = 0; j < 4; ++j) {
    const int col = n0 + wn * 64 + j * 16 + lrow;
    const float bv = bias[col];
    #pragma unroll
    for (int i = 0; i < 2; ++i) {
      #pragma unroll
      for (int r = 0; r < 8; ++r) {
        const int row = m0 + wm * 32 + i * 16 + lhi * 8 + r;
        float val = acc[i][j][r] + bv;
        if (ACT == 1) val = gelu_tanh(val);
        if (OUT_BF16) Yb[(size_t)row * N + col] = (__bf16)val;
        else          Yf[(size_t)row * N + col] = val;
      }
    }
  }
}

// ---------------------------------------------------------------------------
// Flash attention, bf16 Q/K/V/ctx in (B,S,DM) layout (head h = cols h*64..+64).
// One wave per (16-query tile, head, batch); streaming softmax over 32 keys.
// V chunk async-staged to LDS; P round-trips LDS for C-layout -> A-layout.
// ---------------------------------------------------------------------------
__global__ __launch_bounds__(32) void attention_kernel(
    const __bf16* __restrict__ q, const __bf16* __restrict__ k,
    const __bf16* __restrict__ v, const unsigned char* __restrict__ mask,
    __bf16* __restrict__ ctx) {
  const int qb = blockIdx.x, h = blockIdx.y, b = blockIdx.z;
  const int lane = threadIdx.x;
  const int lrow = lane & 15;
  const int lhi  = lane >> 4;
  const int kbA  = lhi * 16;
  __shared__ __align__(16) __bf16 sP[16 * 32];
  __shared__ __align__(16) __bf16 sV[32 * 64];

  const size_t hoff = (size_t)b * S_ * DM_ + (size_t)h * HD_;
  const __bf16* qp = q + hoff;
  const __bf16* kp = k + hoff;
  const __bf16* vp = v + hoff;
  const unsigned char* mp = mask + (size_t)b * S_ * S_;
  const float NEG_INF = -__builtin_huge_valf();

  // Q A-fragments: 16x64 = two K-chunks of 32, direct b128 bf16 loads.
  Frag16 aq[2];
  const int srow = qb * 16 + lrow;
  #pragma unroll
  for (int c = 0; c < 2; ++c) {
    const __bf16* p0 = qp + (size_t)srow * DM_ + c * 32 + kbA;
    aq[c].h8[0] = *(const bf16x8*)p0;
    aq[c].h8[1] = *(const bf16x8*)(p0 + 8);
  }

  float mrow[8], lsum[8];
  #pragma unroll
  for (int r = 0; r < 8; ++r) { mrow[r] = NEG_INF; lsum[r] = 0.0f; }
  v8f o[4] = {};

  const int myrow0 = qb * 16 + lhi * 8;

  for (int jj = 0; jj < S_; jj += 32) {
    // Stage V chunk (32 keys x 64 dims bf16 = 4KB) to LDS, async if possible.
    #pragma unroll
    for (int i = 0; i < 8; ++i) {
      const int c0 = lane + i * 32;          // 0..255 b128 chunks
      const int row = c0 >> 3, ch = (c0 & 7) * 8;
      copy_b128_g2l(vp + (size_t)(jj + row) * DM_ + ch, &sV[row * 64 + ch]);
    }

    // Scores: two 16x16 tiles over keys jj..jj+31 (Q.K^T, K frags direct).
    float sc[2][8];
    #pragma unroll
    for (int t = 0; t < 2; ++t) {
      v8f s = {0.f, 0.f, 0.f, 0.f, 0.f, 0.f, 0.f, 0.f};
      #pragma unroll
      for (int c = 0; c < 2; ++c) {
        Frag16 bk;
        const __bf16* kq =
            kp + (size_t)(jj + t * 16 + lrow) * DM_ + c * 32 + lhi * 16;
        bk.h8[0] = *(const bf16x8*)kq;
        bk.h8[1] = *(const bf16x8*)(kq + 8);
        s = __builtin_amdgcn_wmma_f32_16x16x32_bf16(
            false, aq[c].v, false, bk.v, (short)0, s, false, false);
      }
      const int kcol = jj + t * 16 + lrow;
      #pragma unroll
      for (int r = 0; r < 8; ++r) {
        float val = s[r] * 0.125f;  // 1/sqrt(HD)
        if (mp[(size_t)(myrow0 + r) * S_ + kcol]) val = -1e32f;
        sc[t][r] = val;
      }
    }

    // Streaming softmax: row reductions across the 16 lanes of the half-wave.
    #pragma unroll
    for (int r = 0; r < 8; ++r) {
      float mx = fmaxf(sc[0][r], sc[1][r]);
      #pragma unroll
      for (int off = 8; off > 0; off >>= 1)
        mx = fmaxf(mx, __shfl_xor(mx, off, 16));
      const float mn = fmaxf(mrow[r], mx);
      const float alpha = (mrow[r] == NEG_INF) ? 0.0f : __expf(mrow[r] - mn);
      const float p0 = __expf(sc[0][r] - mn);
      const float p1 = __expf(sc[1][r] - mn);
      float rs = p0 + p1;
      #pragma unroll
      for (int off = 8; off > 0; off >>= 1)
        rs += __shfl_xor(rs, off, 16);
      lsum[r] = lsum[r] * alpha + rs;
      mrow[r] = mn;
      #pragma unroll
      for (int dt = 0; dt < 4; ++dt) o[dt][r] *= alpha;
      sP[(lhi * 8 + r) * 32 + lrow]      = (__bf16)p0;
      sP[(lhi * 8 + r) * 32 + 16 + lrow] = (__bf16)p1;
    }

    wait_async_lds();  // sV resident before ds reads

    // O += P(16x32) @ V(32x64)
    Frag16 ap;
    ap.h8[0] = *(const bf16x8*)&sP[lrow * 32 + kbA];
    ap.h8[1] = *(const bf16x8*)&sP[lrow * 32 + kbA + 8];
    #pragma unroll
    for (int dt = 0; dt < 4; ++dt) {
      Frag16 bvf;
      #pragma unroll
      for (int e = 0; e < 16; ++e)
        bvf.v[e] = sV[(lhi * 16 + e) * 64 + dt * 16 + lrow];
      o[dt] = __builtin_amdgcn_wmma_f32_16x16x32_bf16(
          false, ap.v, false, bvf.v, (short)0, o[dt], false, false);
    }
  }

  #pragma unroll
  for (int dt = 0; dt < 4; ++dt) {
    #pragma unroll
    for (int r = 0; r < 8; ++r) {
      const float denom = lsum[r];
      const float val = (denom > 0.0f) ? o[dt][r] / denom : 0.0f;
      ctx[hoff + (size_t)(qb * 16 + lhi * 8 + r) * DM_ + dt * 16 + lrow] =
          (__bf16)val;
    }
  }
}

// ---------------------------------------------------------------------------
// out = LayerNorm(a + b)*g + beta ; optional bf16 copy for downstream GEMM A.
// ---------------------------------------------------------------------------
template <int WRITE_B>
__global__ __launch_bounds__(256) void add_layernorm(
    const float* __restrict__ a, const float* __restrict__ bsrc,
    const float* __restrict__ g, const float* __restrict__ beta,
    float* __restrict__ outf, __bf16* __restrict__ outb) {
  __shared__ float sh_s[8];
  __shared__ float sh_q[8];
  const int row = blockIdx.x;
  const int tid = threadIdx.x;
  const float* pa = a + (size_t)row * DM_;
  const float* pb = bsrc + (size_t)row * DM_;

  float rv[4];
  float s = 0.f, qq = 0.f;
  #pragma unroll
  for (int i = 0; i < 4; ++i) {
    const int c = tid + i * 256;
    rv[i] = pa[c] + pb[c];
    s += rv[i];
    qq += rv[i] * rv[i];
  }
  #pragma unroll
  for (int off = 16; off > 0; off >>= 1) {
    s  += __shfl_xor(s, off);
    qq += __shfl_xor(qq, off);
  }
  const int wave = tid >> 5, lane = tid & 31;
  if (lane == 0) { sh_s[wave] = s; sh_q[wave] = qq; }
  __syncthreads();
  s = 0.f; qq = 0.f;
  #pragma unroll
  for (int w = 0; w < 8; ++w) { s += sh_s[w]; qq += sh_q[w]; }

  const float mu = s * (1.0f / DM_);
  const float var = qq * (1.0f / DM_) - mu * mu;
  const float rstd = rsqrtf(var + EPS_);
  #pragma unroll
  for (int i = 0; i < 4; ++i) {
    const int c = tid + i * 256;
    const float val = (rv[i] - mu) * rstd * g[c] + beta[c];
    outf[(size_t)row * DM_ + c] = val;
    if (WRITE_B) outb[(size_t)row * DM_ + c] = (__bf16)val;
  }
}

// ---------------------------------------------------------------------------
extern "C" void kernel_launch(void* const* d_in, const int* in_sizes, int n_in,
                              void* d_out, int out_size, void* d_ws, size_t ws_size,
                              hipStream_t stream) {
  (void)in_sizes; (void)n_in; (void)out_size; (void)ws_size;

  const float* x   = (const float*)d_in[0];
  const unsigned char* mask = (const unsigned char*)d_in[1];
  const float* Wq  = (const float*)d_in[2];
  const float* bq  = (const float*)d_in[3];
  const float* Wk  = (const float*)d_in[4];
  const float* bk  = (const float*)d_in[5];
  const float* Wv  = (const float*)d_in[6];
  const float* bv  = (const float*)d_in[7];
  const float* Wo  = (const float*)d_in[8];
  const float* bo  = (const float*)d_in[9];
  const float* g1  = (const float*)d_in[10];
  const float* be1 = (const float*)d_in[11];
  const float* W1  = (const float*)d_in[12];
  const float* b1  = (const float*)d_in[13];
  const float* W2  = (const float*)d_in[14];
  const float* b2  = (const float*)d_in[15];
  const float* g2  = (const float*)d_in[16];
  const float* be2 = (const float*)d_in[17];
  float* out = (float*)d_out;

  const int MT = B_ * S_;  // 8192 token rows

  // Workspace carve-up (~280 MB)
  char* p = (char*)d_ws;
  auto carve = [&](size_t bytes) { char* r = p; p += (bytes + 255) & ~(size_t)255; return r; };
  __bf16* xb   = (__bf16*)carve((size_t)MT * DM_ * 2);
  __bf16* Wqb  = (__bf16*)carve((size_t)DM_ * DM_ * 2);
  __bf16* Wkb  = (__bf16*)carve((size_t)DM_ * DM_ * 2);
  __bf16* Wvb  = (__bf16*)carve((size_t)DM_ * DM_ * 2);
  __bf16* Wob  = (__bf16*)carve((size_t)DM_ * DM_ * 2);
  __bf16* W1b  = (__bf16*)carve((size_t)FF_ * DM_ * 2);   // (FF,DM) = W1^T
  __bf16* W2b  = (__bf16*)carve((size_t)DM_ * FF_ * 2);   // (DM,FF) = W2^T
  __bf16* qb   = (__bf16*)carve((size_t)MT * DM_ * 2);
  __bf16* kb_  = (__bf16*)carve((size_t)MT * DM_ * 2);
  __bf16* vb_  = (__bf16*)carve((size_t)MT * DM_ * 2);
  __bf16* ctxb = (__bf16*)carve((size_t)MT * DM_ * 2);
  __bf16* x1b  = (__bf16*)carve((size_t)MT * DM_ * 2);
  __bf16* midb = (__bf16*)carve((size_t)MT * FF_ * 2);
  float*  attn_out = (float*)carve((size_t)MT * DM_ * 4);
  float*  x1f      = (float*)carve((size_t)MT * DM_ * 4);
  float*  ffn_out  = (float*)carve((size_t)MT * DM_ * 4);

  const dim3 blk(256);

  // One-time precision/layout prep: x -> bf16; weights -> bf16 transposed (N,K)
  to_bf16_kernel<<<dim3((MT * DM_) / 1024), blk, 0, stream>>>(x, xb);
  transpose_to_bf16<<<dim3(DM_ / 32, DM_ / 32), blk, 0, stream>>>(Wq, Wqb, DM_, DM_);
  transpose_to_bf16<<<dim3(DM_ / 32, DM_ / 32), blk, 0, stream>>>(Wk, Wkb, DM_, DM_);
  transpose_to_bf16<<<dim3(DM_ / 32, DM_ / 32), blk, 0, stream>>>(Wv, Wvb, DM_, DM_);
  transpose_to_bf16<<<dim3(DM_ / 32, DM_ / 32), blk, 0, stream>>>(Wo, Wob, DM_, DM_);
  transpose_to_bf16<<<dim3(FF_ / 32, DM_ / 32), blk, 0, stream>>>(W1, W1b, DM_, FF_);
  transpose_to_bf16<<<dim3(DM_ / 32, FF_ / 32), blk, 0, stream>>>(W2, W2b, FF_, DM_);

  const dim3 gProj(MT / 128, DM_ / 128);
  const dim3 gFfn1(MT / 128, FF_ / 128);

  // QKV projections (bf16 in/out, f32 accumulate)
  gemm_bf16<0, 1><<<gProj, blk, 0, stream>>>(xb, Wqb, bq, nullptr, qb,  MT, DM_, DM_);
  gemm_bf16<0, 1><<<gProj, blk, 0, stream>>>(xb, Wkb, bk, nullptr, kb_, MT, DM_, DM_);
  gemm_bf16<0, 1><<<gProj, blk, 0, stream>>>(xb, Wvb, bv, nullptr, vb_, MT, DM_, DM_);

  // Flash attention
  attention_kernel<<<dim3(S_ / 16, H_, B_), dim3(32), 0, stream>>>(
      qb, kb_, vb_, mask, ctxb);

  // Output projection + residual LN (writes f32 residual + bf16 GEMM input)
  gemm_bf16<0, 0><<<gProj, blk, 0, stream>>>(ctxb, Wob, bo, attn_out, nullptr, MT, DM_, DM_);
  add_layernorm<1><<<dim3(MT), blk, 0, stream>>>(x, attn_out, g1, be1, x1f, x1b);

  // FFN: GELU(x1 @ W1 + b1) @ W2 + b2, then residual LN -> out
  gemm_bf16<1, 1><<<gFfn1, blk, 0, stream>>>(x1b, W1b, b1, nullptr, midb, MT, FF_, DM_);
  gemm_bf16<0, 0><<<gProj, blk, 0, stream>>>(midb, W2b, b2, ffn_out, nullptr, MT, DM_, FF_);
  add_layernorm<0><<<dim3(MT), blk, 0, stream>>>(x1f, ffn_out, g2, be2, out, nullptr);
}
